// TimestampGuidedAttention_17248588661225
// MI455X (gfx1250) — matrix operand order repaired and checked by reading, not compiled, not run on hardware
//
#include <hip/hip_runtime.h>
#include <math.h>
#include <stdint.h>

// Problem constants (from reference): B=32, T=2048, H=512
#define BB 32
#define TT 2048
#define HH 512

// LDS B-tile row pitch in bytes: 64 data bytes (32 bf16) + 16 pad.
// Per-row bank offset = 20 dwords; 20*r mod 64 is a permutation over the 16
// rows a b128 read-group touches -> conflict-free ds_load_b128 column reads.
#define BPITCH 80

// ---------------------------------------------------------------------------
// Types for WMMA bf16 path (v_wmma_f32_16x16x32_bf16)
// ---------------------------------------------------------------------------
typedef __bf16 bf16_t;
typedef bf16_t v16bf __attribute__((ext_vector_type(16)));
typedef float  v8f   __attribute__((ext_vector_type(8)));

union FragBF {
    v16bf v;
    uint4 q[2];   // 2 x 16B = 32B = 16 bf16 per lane
};

__device__ __forceinline__ unsigned short f32_to_bf16_bits(float f) {
    unsigned int u = __float_as_uint(f);
    u += 0x7FFFu + ((u >> 16) & 1u);      // round to nearest even
    return (unsigned short)(u >> 16);
}

// ---------------------------------------------------------------------------
// gfx1250 async copy: global -> LDS, 16 bytes per lane, tracked by ASYNCcnt.
// VDST = LDS byte offset (low 32 bits of the generic LDS pointer), VADDR =
// 64-bit global address. (CDNA5 ISA)
// ---------------------------------------------------------------------------
__device__ __forceinline__ void async_copy_b128(void* lds_ptr, const void* gptr) {
    unsigned loff = (unsigned)(uintptr_t)lds_ptr;   // flat LDS addr[31:0] = LDS offset
    asm volatile("global_load_async_to_lds_b128 %0, %1, off"
                 :: "v"(loff), "v"(gptr)
                 : "memory");
}
__device__ __forceinline__ void wait_async_zero() {
    asm volatile("s_wait_asynccnt 0" ::: "memory");
}

// A-matrix 16x32 bf16 fragment (row-major source, leading dim = ld)
// ISA layout: lanes 0-15 hold row M=lane, K = {0..7, 16..23};
//             lanes 16-31 hold row M=lane-16, K = {8..15, 24..31}.
__device__ __forceinline__ void load_a(FragBF& f, const unsigned short* base,
                                       int ld, int row, int k0, int hi) {
    const unsigned short* p = base + (size_t)row * ld + (k0 + hi * 8);
    f.q[0] = *(const uint4*)(p);        // K = k0+hi*8 .. +7      -> elems 0..7
    f.q[1] = *(const uint4*)(p + 16);   // K = k0+hi*8+16 .. +23  -> elems 8..15
}

// B-matrix 32x16 bf16 fragment read from an LDS-staged tile. Tile row = output
// column s (pre-transposed kernel), 32 K-values (64B) per row, BPITCH stride.
// ISA layout: lanes 0-15 hold column N=lane with K={0..7, 16..23}; lanes 16-31
// hold column N=lane-16 with K={8..15, 24..31}.
__device__ __forceinline__ void load_b_lds(FragBF& f, const unsigned char* sh,
                                           int col, int hi) {
    const unsigned char* p = sh + col * BPITCH + hi * 16;
    f.q[0] = *(const uint4*)(p);        // K = hi*8 + 0..7
    f.q[1] = *(const uint4*)(p + 32);   // K = hi*8 + 16..23
}

__device__ __forceinline__ v8f wmma_bf16(const FragBF& a, const FragBF& b, v8f c) {
    return __builtin_amdgcn_wmma_f32_16x16x32_bf16(
        /*neg_a=*/false, a.v, /*neg_b=*/false, b.v,
        /*c_mod=*/(short)0, c, /*reuse_a=*/false, /*reuse_b=*/false);
}

// Stage one 64(N) x 32(K) bf16 B-tile (4KB) into LDS: 256 chunks of 16B,
// exactly one async transfer per thread.
__device__ __forceinline__ void stage_b_tile(unsigned char* dst,
                                             const unsigned short* srcT,
                                             int n0, int k0, int tid) {
    const int row = tid >> 2;     // 0..63  (output column s within block)
    const int c   = tid & 3;      // 16B chunk within the 64B row
    async_copy_b128(dst + row * BPITCH + c * 16,
                    srcT + (size_t)(n0 + row) * TT + k0 + c * 8);
}

// ---------------------------------------------------------------------------
// Kernel 1: tiled transpose + f32->bf16 cast.
//   in : [R, C] f32 row-major (per batch, batch stride in_bs elements)
//   out: [C, R] bf16 row-major (per batch, batch stride out_bs elements)
// ---------------------------------------------------------------------------
__global__ void __launch_bounds__(256)
transpose_cast_kernel(const float* __restrict__ in, unsigned short* __restrict__ out,
                      int R, int C, size_t in_bs, size_t out_bs) {
    __shared__ float tile[32][33];
    const int bz = blockIdx.z;
    const float* ip = in + (size_t)bz * in_bs;
    unsigned short* op = out + (size_t)bz * out_bs;
    const int r0 = blockIdx.y * 32;
    const int c0 = blockIdx.x * 32;
    for (int i = threadIdx.y; i < 32; i += 8)
        tile[i][threadIdx.x] = ip[(size_t)(r0 + i) * C + c0 + threadIdx.x];
    __syncthreads();
    for (int i = threadIdx.y; i < 32; i += 8)
        op[(size_t)(c0 + i) * R + r0 + threadIdx.x] =
            f32_to_bf16_bits(tile[threadIdx.x][i]);
}

// ---------------------------------------------------------------------------
// Kernel 2: fused GEMM1+GEMM2 + tanh + lambda-combine -> delta (bf16).
//   gamma = tanh(xp @ Kx + bx); beta = tanh(tp @ Kt + bt)
//   delta = lam*gamma + (1-lam)*beta, lam indexed by h = m % H
// Block: 256 threads (8 wave32), block tile 128(M) x 64(N), wave tile 32x32.
// B tiles double-buffered in LDS via gfx1250 async-to-LDS, overlapped with WMMA.
// ---------------------------------------------------------------------------
__global__ void __launch_bounds__(256)
gemm12_fused_kernel(const unsigned short* __restrict__ xp,
                    const unsigned short* __restrict__ tp,
                    const unsigned short* __restrict__ kxT,
                    const unsigned short* __restrict__ ktT,
                    const float* __restrict__ bias_x,
                    const float* __restrict__ bias_t,
                    const float* __restrict__ lambd,
                    unsigned short* __restrict__ delta) {
    __shared__ __align__(16) unsigned char shB[2][2][64 * BPITCH]; // [buf][kx/kt]

    const int tid  = threadIdx.x;
    const int lane = tid & 31;
    const int wave = tid >> 5;
    const int hi = lane >> 4;       // K-half select
    const int ln = lane & 15;       // row (A) / col (B,C,D) within tile
    const int n0 = blockIdx.x * 64;
    const int m_blk = blockIdx.y * 128 + (wave >> 1) * 32;
    const int wn32 = (wave & 1) * 32;

    v8f ag[2][2] = {};   // gamma accumulators
    v8f ab[2][2] = {};   // beta accumulators

    // Prologue: stage K-step 0 into buffer 0.
    stage_b_tile(shB[0][0], kxT, n0, 0, tid);
    stage_b_tile(shB[0][1], ktT, n0, 0, tid);
    wait_async_zero();
    __syncthreads();

    int p = 0;
    for (int k0 = 0; k0 < TT; k0 += 32) {
        // Prefetch next K-step's B tiles into the other buffer (overlaps WMMA).
        if (k0 + 32 < TT) {
            stage_b_tile(shB[p ^ 1][0], kxT, n0, k0 + 32, tid);
            stage_b_tile(shB[p ^ 1][1], ktT, n0, k0 + 32, tid);
        }

        FragBF ax[2], at[2], bx[2], bt[2];
#pragma unroll
        for (int i = 0; i < 2; ++i) {
            load_a(ax[i], xp, TT, m_blk + 16 * i + ln, k0, hi);
            load_a(at[i], tp, TT, m_blk + 16 * i + ln, k0, hi);
            load_b_lds(bx[i], shB[p][0], wn32 + 16 * i + ln, hi);
            load_b_lds(bt[i], shB[p][1], wn32 + 16 * i + ln, hi);
        }
#pragma unroll
        for (int i = 0; i < 2; ++i)
#pragma unroll
            for (int j = 0; j < 2; ++j) {
                ag[i][j] = wmma_bf16(ax[i], bx[j], ag[i][j]);
                ab[i][j] = wmma_bf16(at[i], bt[j], ab[i][j]);
            }

        wait_async_zero();   // next buffer landed
        __syncthreads();     // everyone done reading buffer p
        p ^= 1;
    }

    // Epilogue. D layout: VGPR r -> row M = r + 8*hi (within 16x16), col N = ln.
#pragma unroll
    for (int i = 0; i < 2; ++i) {
#pragma unroll
        for (int j = 0; j < 2; ++j) {
            const int s = n0 + wn32 + 16 * j + ln;
            const float bxv = bias_x[s];
            const float btv = bias_t[s];
#pragma unroll
            for (int r = 0; r < 8; ++r) {
                const int m = m_blk + 16 * i + r + 8 * hi;   // m = b*H + h
                const float lam = lambd[m & (HH - 1)];
                const float g = tanhf(ag[i][j][r] + bxv);
                const float be = tanhf(ab[i][j][r] + btv);
                const float d = lam * g + (1.0f - lam) * be;
                delta[(size_t)m * TT + s] = f32_to_bf16_bits(d);
            }
        }
    }
}

// ---------------------------------------------------------------------------
// Kernel 3: GEMM3: logits = delta @ Ka  (fp32 out, softmax done separately)
// Same blocking; single B matrix staged through the async LDS pipeline.
// ---------------------------------------------------------------------------
__global__ void __launch_bounds__(256)
gemm3_kernel(const unsigned short* __restrict__ delta,
             const unsigned short* __restrict__ kaT,
             float* __restrict__ logits) {
    __shared__ __align__(16) unsigned char shB[2][64 * BPITCH];

    const int tid  = threadIdx.x;
    const int lane = tid & 31;
    const int wave = tid >> 5;
    const int hi = lane >> 4;
    const int ln = lane & 15;
    const int n0 = blockIdx.x * 64;
    const int m_blk = blockIdx.y * 128 + (wave >> 1) * 32;
    const int wn32 = (wave & 1) * 32;

    v8f acc[2][2] = {};

    stage_b_tile(shB[0], kaT, n0, 0, tid);
    wait_async_zero();
    __syncthreads();

    int p = 0;
    for (int k0 = 0; k0 < TT; k0 += 32) {
        if (k0 + 32 < TT)
            stage_b_tile(shB[p ^ 1], kaT, n0, k0 + 32, tid);

        FragBF a[2], b[2];
#pragma unroll
        for (int i = 0; i < 2; ++i) {
            load_a(a[i], delta, TT, m_blk + 16 * i + ln, k0, hi);
            load_b_lds(b[i], shB[p], wn32 + 16 * i + ln, hi);
        }
#pragma unroll
        for (int i = 0; i < 2; ++i)
#pragma unroll
            for (int j = 0; j < 2; ++j)
                acc[i][j] = wmma_bf16(a[i], b[j], acc[i][j]);

        wait_async_zero();
        __syncthreads();
        p ^= 1;
    }

#pragma unroll
    for (int i = 0; i < 2; ++i)
#pragma unroll
        for (int j = 0; j < 2; ++j) {
            const int s = n0 + wn32 + 16 * j + ln;
#pragma unroll
            for (int r = 0; r < 8; ++r) {
                const int m = m_blk + 16 * i + r + 8 * hi;
                logits[(size_t)m * TT + s] = acc[i][j][r];
            }
        }
}

// ---------------------------------------------------------------------------
// Kernel 4: row softmax over T + transpose back [B,H,T] -> [B,T,H]
// One 256-thread block per (b,h) row; 8 elements per thread held in registers.
// ---------------------------------------------------------------------------
__global__ void __launch_bounds__(256)
softmax_transpose_kernel(const float* __restrict__ logits, float* __restrict__ out) {
    const int row = blockIdx.x;              // row = b*H + h
    const int b = row >> 9;                  // / HH
    const int h = row & (HH - 1);
    const float* lp = logits + (size_t)row * TT;
    const int tid = threadIdx.x;

    __shared__ float sbuf[256];

    float vals[8];
    float m = -3.402823e38f;
#pragma unroll
    for (int i = 0; i < 8; ++i) {
        vals[i] = lp[tid + 256 * i];
        m = fmaxf(m, vals[i]);
    }
    sbuf[tid] = m;
    __syncthreads();
    for (int s = 128; s > 0; s >>= 1) {
        if (tid < s) sbuf[tid] = fmaxf(sbuf[tid], sbuf[tid + s]);
        __syncthreads();
    }
    m = sbuf[0];
    __syncthreads();

    float sum = 0.0f;
#pragma unroll
    for (int i = 0; i < 8; ++i) {
        vals[i] = __expf(vals[i] - m);
        sum += vals[i];
    }
    sbuf[tid] = sum;
    __syncthreads();
    for (int s = 128; s > 0; s >>= 1) {
        if (tid < s) sbuf[tid] += sbuf[tid + s];
        __syncthreads();
    }
    const float inv = 1.0f / sbuf[0];

#pragma unroll
    for (int i = 0; i < 8; ++i) {
        const int s = tid + 256 * i;
        out[(size_t)b * TT * HH + (size_t)s * HH + h] = vals[i] * inv;
    }
}

// ---------------------------------------------------------------------------
// Host-side launcher
// Workspace layout (bytes):
//   [0,   64Mi)  xp  bf16  [B*H, T]      (later aliased by fp32 logits)
//   [64Mi,128Mi) tp  bf16  [B*H, T]      (aliased by logits too)
//   [128Mi,192Mi) delta bf16 [B*H, T]
//   [192Mi,200Mi) KxT bf16 [T, T]
//   [200Mi,208Mi) KtT bf16 [T, T]
//   [208Mi,216Mi) KaT bf16 [T, T]
//   logits fp32 [B*H, T] = 128 MiB aliases xp+tp (dead by GEMM3 time)
// ---------------------------------------------------------------------------
extern "C" void kernel_launch(void* const* d_in, const int* in_sizes, int n_in,
                              void* d_out, int out_size, void* d_ws, size_t ws_size,
                              hipStream_t stream) {
    const float* x      = (const float*)d_in[0];
    const float* t      = (const float*)d_in[1];
    const float* kx     = (const float*)d_in[2];
    const float* kt     = (const float*)d_in[3];
    const float* ka     = (const float*)d_in[4];
    const float* bias_x = (const float*)d_in[5];
    const float* bias_t = (const float*)d_in[6];
    const float* lambd  = (const float*)d_in[7];
    float* out = (float*)d_out;

    char* ws = (char*)d_ws;
    const size_t MB = 1024ull * 1024ull;
    unsigned short* xp    = (unsigned short*)(ws + 0);
    unsigned short* tp    = (unsigned short*)(ws + 64 * MB);
    unsigned short* delta = (unsigned short*)(ws + 128 * MB);
    unsigned short* kxT   = (unsigned short*)(ws + 192 * MB);
    unsigned short* ktT   = (unsigned short*)(ws + 200 * MB);
    unsigned short* kaT   = (unsigned short*)(ws + 208 * MB);
    float* logits         = (float*)(ws + 0);   // aliases xp+tp (dead by then)

    dim3 tb(32, 8, 1);

    // xp[b*H+h][t] = x[b][t][h]  (per-batch [T,H] -> [H,T] transpose + cast)
    transpose_cast_kernel<<<dim3(HH / 32, TT / 32, BB), tb, 0, stream>>>(
        x, xp, TT, HH, (size_t)TT * HH, (size_t)HH * TT);
    transpose_cast_kernel<<<dim3(HH / 32, TT / 32, BB), tb, 0, stream>>>(
        t, tp, TT, HH, (size_t)TT * HH, (size_t)HH * TT);
    // kT[s][k] = kernel[k][s]
    transpose_cast_kernel<<<dim3(TT / 32, TT / 32, 1), tb, 0, stream>>>(
        kx, kxT, TT, TT, 0, 0);
    transpose_cast_kernel<<<dim3(TT / 32, TT / 32, 1), tb, 0, stream>>>(
        kt, ktT, TT, TT, 0, 0);
    transpose_cast_kernel<<<dim3(TT / 32, TT / 32, 1), tb, 0, stream>>>(
        ka, kaT, TT, TT, 0, 0);

    // Fused gamma/beta GEMMs + tanh + lambda combine -> delta
    dim3 gg(TT / 64, (BB * HH) / 128, 1);   // 32 x 128 blocks
    gemm12_fused_kernel<<<gg, 256, 0, stream>>>(xp, tp, kxT, ktT,
                                                bias_x, bias_t, lambd, delta);

    // logits = delta @ Ka
    gemm3_kernel<<<gg, 256, 0, stream>>>(delta, kaT, logits);

    // softmax over T, transpose back to [B, T, H]
    softmax_transpose_kernel<<<BB * HH, 256, 0, stream>>>(logits, out);
}
